// GATv2Layer_70007966925420
// MI455X (gfx1250) — compile-verified, matrix operand order checked by reading, not run on hardware
//
#include <hip/hip_runtime.h>

// ---------------------------------------------------------------------------
// GATv2 (broadcast-score variant) on MI455X / gfx1250
//   Wh = X @ W.T ; e = relu(Wh) @ a_w ; P = exp(e - max e)
//   out = relu( (A @ [diag(P)Wh | P]) split -> Num / Den )
// Main cost: streaming A (256 MB) once -> bf16 WMMA (v_wmma_f32_16x16x32_bf16)
// ---------------------------------------------------------------------------

typedef __attribute__((ext_vector_type(16))) __bf16 v16bf;
typedef __attribute__((ext_vector_type(8)))  __bf16 v8bf;
typedef __attribute__((ext_vector_type(8)))  float  v8f;
typedef __attribute__((ext_vector_type(4)))  float  v4f;

#define GAT_N      8192
#define GAT_INDIM  256
#define GAT_OUTDIM 128
#define GAT_YCOLS  144   // 128 Wh cols + 1 denominator col + 15 pad (9 N-tiles of 16)

// ---------------- Kernel 1: Wh = X @ W.T and e = relu(Wh) @ a_w ------------
__global__ __launch_bounds__(128)
void gat_wh_e_kernel(const float* __restrict__ X, const float* __restrict__ W,
                     const float* __restrict__ aw, float* __restrict__ Wh,
                     float* __restrict__ e) {
    __shared__ float xs[GAT_INDIM];
    __shared__ float red[128];
    const int i = blockIdx.x;       // row of X
    const int d = threadIdx.x;      // output feature 0..127
    xs[d]       = X[(size_t)i * GAT_INDIM + d];
    xs[d + 128] = X[(size_t)i * GAT_INDIM + 128 + d];
    __syncthreads();
    const float* wrow = W + (size_t)d * GAT_INDIM;
    float acc = 0.f;
#pragma unroll 8
    for (int k = 0; k < GAT_INDIM; ++k) acc = fmaf(xs[k], wrow[k], acc);
    Wh[(size_t)i * GAT_OUTDIM + d] = acc;
    red[d] = fmaxf(acc, 0.f) * aw[d];
    __syncthreads();
    for (int s = 64; s > 0; s >>= 1) {
        if (d < s) red[d] += red[d + s];
        __syncthreads();
    }
    if (d == 0) e[i] = red[0];
}

// ---------------- Kernel 2: global max of e --------------------------------
__global__ __launch_bounds__(256)
void gat_max_kernel(const float* __restrict__ e, float* __restrict__ maxe) {
    __shared__ float red[256];
    float m = -3.0e38f;
    for (int j = threadIdx.x; j < GAT_N; j += 256) m = fmaxf(m, e[j]);
    red[threadIdx.x] = m;
    __syncthreads();
    for (int s = 128; s > 0; s >>= 1) {
        if (threadIdx.x < s) red[threadIdx.x] = fmaxf(red[threadIdx.x], red[threadIdx.x + s]);
        __syncthreads();
    }
    if (threadIdx.x == 0) *maxe = red[0];
}

// ---------------- Kernel 3: Yt[d][j] (transposed, bf16) --------------------
// Yt rows 0..127: P[j]*Wh[j][d]; row 128: P[j]; rows 129..143: 0 (pad tile)
__global__ __launch_bounds__(256)
void gat_build_yt_kernel(const float* __restrict__ Wh, const float* __restrict__ e,
                         const float* __restrict__ maxe, __bf16* __restrict__ Yt) {
    const int j = blockIdx.x * 256 + threadIdx.x;   // 0..8191
    const float p = __expf(e[j] - *maxe);
    const float* wr = Wh + (size_t)j * GAT_OUTDIM;
#pragma unroll 4
    for (int d = 0; d < GAT_OUTDIM; ++d)
        Yt[(size_t)d * GAT_N + j] = (__bf16)(p * wr[d]);
    Yt[(size_t)128 * GAT_N + j] = (__bf16)p;
#pragma unroll
    for (int d = 129; d < GAT_YCOLS; ++d)
        Yt[(size_t)d * GAT_N + j] = (__bf16)0.f;
}

// ---------------- Kernel 4: main GEMM  A[8192x8192] @ Y[8192x144] ----------
// 4 waves/block; each wave owns one 16-row M-tile x nine 16-col N-tiles.
__global__ __launch_bounds__(128)
void gat_gemm_kernel(const float* __restrict__ A, const __bf16* __restrict__ Yt,
                     float* __restrict__ out) {
    const int lane  = threadIdx.x & 31;
    const int wave  = threadIdx.x >> 5;
    const int mtile = blockIdx.x * 4 + wave;      // 0..511
    const int m     = lane & 15;                  // row within tile / col within B tile
    const int h     = lane >> 4;                  // K-half selector per ISA operand layout
    const float* arow = A + (size_t)(mtile * 16 + m) * GAT_N;

    v8f acc[9];
#pragma unroll
    for (int t = 0; t < 9; ++t) { v8f z = {}; acc[t] = z; }

    for (int k0 = 0; k0 < GAT_N; k0 += 32) {
        // speculative prefetch of the A stream (global_prefetch_b8)
        __builtin_prefetch(arow + k0 + 256, 0, 0);

        // A operand: lane holds K = k0+8h+0..7 and k0+16+8h+0..7 (f32 -> bf16)
        v4f a0 = *(const v4f*)(arow + k0 + 8 * h);
        v4f a1 = *(const v4f*)(arow + k0 + 8 * h + 4);
        v4f a2 = *(const v4f*)(arow + k0 + 16 + 8 * h);
        v4f a3 = *(const v4f*)(arow + k0 + 16 + 8 * h + 4);
        v16bf aop;
#pragma unroll
        for (int i = 0; i < 4; ++i) {
            aop[i]      = (__bf16)a0[i];
            aop[4 + i]  = (__bf16)a1[i];
            aop[8 + i]  = (__bf16)a2[i];
            aop[12 + i] = (__bf16)a3[i];
        }

#pragma unroll
        for (int t = 0; t < 9; ++t) {
            // B operand: column n = m of tile t is row (t*16+m) of transposed Y
            const __bf16* bcol = Yt + (size_t)(t * 16 + m) * GAT_N + k0 + 8 * h;
            v8bf blo = *(const v8bf*)bcol;
            v8bf bhi = *(const v8bf*)(bcol + 16);
            v16bf bop;
#pragma unroll
            for (int i = 0; i < 8; ++i) { bop[i] = blo[i]; bop[8 + i] = bhi[i]; }
            acc[t] = __builtin_amdgcn_wmma_f32_16x16x32_bf16(
                false, aop, false, bop, (short)0, acc[t], false, false);
        }
    }

    // C/D layout: VGPR v, lane L -> (row = v + 8*(L>>4), col = L&15).
    // Denominator = tile 8, local col 0 -> lanes 0 / 16 hold it for each half.
#pragma unroll
    for (int v = 0; v < 8; ++v) {
        float den = __shfl(acc[8][v], lane & 16, 32);
        float inv = den > 0.f ? 1.f / den : 0.f;
        const int orow = mtile * 16 + v + 8 * h;
#pragma unroll
        for (int t = 0; t < 8; ++t) {
            float val = acc[t][v] * inv;
            out[(size_t)orow * GAT_OUTDIM + t * 16 + (lane & 15)] = fmaxf(val, 0.f);
        }
    }
}

// ---------------------------------------------------------------------------
extern "C" void kernel_launch(void* const* d_in, const int* in_sizes, int n_in,
                              void* d_out, int out_size, void* d_ws, size_t ws_size,
                              hipStream_t stream) {
    const float* X  = (const float*)d_in[0];  // [8192, 256]
    const float* A  = (const float*)d_in[1];  // [8192, 8192]
    const float* W  = (const float*)d_in[2];  // [128, 256]
    const float* aw = (const float*)d_in[3];  // [128]
    float* out = (float*)d_out;               // [8192, 128]

    char* ws = (char*)d_ws;
    float*  Wh   = (float*)(ws);                         // 4 MB
    float*  e    = (float*)(ws + 4194304);               // 32 KB
    float*  maxe = (float*)(ws + 4227072);               // 4 B (256-B aligned)
    __bf16* Yt   = (__bf16*)(ws + 4227328);              // 144*8192*2 = 2.25 MB

    gat_wh_e_kernel<<<GAT_N, 128, 0, stream>>>(X, W, aw, Wh, e);
    gat_max_kernel<<<1, 256, 0, stream>>>(e, maxe);
    gat_build_yt_kernel<<<GAT_N / 256, 256, 0, stream>>>(Wh, e, maxe, Yt);
    gat_gemm_kernel<<<(GAT_N / 16) / 4, 128, 0, stream>>>(A, Yt, out);
}
